// Seq2Seq_78546361909919
// MI455X (gfx1250) — matrix operand order repaired and checked by reading, not compiled, not run on hardware
//
#include <hip/hip_runtime.h>
#include <hip/hip_bf16.h>
#include <cstdint>
#include <cstddef>

// ---------------- problem constants ----------------
#define S_SRC  50
#define S_TGT  50
#define NB     64          // batch
#define EDIM   300
#define EPAD   320         // E padded to multiple of 32
#define HDIM   512
#define G4H    2048        // 4*H
#define H2     1024        // 2*H
#define VOCAB  10000

typedef __attribute__((ext_vector_type(16))) _Float16 v16h;
typedef __attribute__((ext_vector_type(8)))  _Float16 v8h;
typedef __attribute__((ext_vector_type(8)))  float    v8f;

#define WMMA_F16(a, b, c) \
  __builtin_amdgcn_wmma_f32_16x16x32_f16(false, (a), false, (b), (short)0, (c), false, false)

// =====================================================================
// 3-pair WMMA GEMM:  C(M x N) = sum_p A_p(M x K_p) * B_p^T + bias
//   A_p row-major (M x K_p), lda = K_p
//   B_p row-major (N x K_p), row stride ldb_p (halves)
//   C   row-major (M x N) f32, optional f16 mirror C16
// Block = 128 threads (4 waves). blockIdx.y selects 64-row M-block,
// blockIdx.x selects 64-col N-block; wave w owns rows [16w,16w+16).
// K_p must be a multiple of 32; M a multiple of 64; N a multiple of 16.
// =====================================================================
__global__ void gemm3_wmma(
    const _Float16* __restrict__ A1, const _Float16* __restrict__ B1, int K1, int ldb1,
    const _Float16* __restrict__ A2, const _Float16* __restrict__ B2, int K2, int ldb2,
    const _Float16* __restrict__ A3, const _Float16* __restrict__ B3, int K3, int ldb3,
    const float* __restrict__ bias,
    float* __restrict__ C, _Float16* __restrict__ C16, int N)
{
  const int lane = threadIdx.x & 31;
  const int wave = threadIdx.x >> 5;
  const int m0   = blockIdx.y * 64 + wave * 16;   // absolute row base of this wave's tile
  const int n0   = blockIdx.x * 64;
  const int r    = lane & 15;
  const bool hi  = (lane >= 16);
  const int aoff = hi ? 16 : 0;   // A frag: lane<16 K 0..15, lane>=16 K 16..31
  const int boff = hi ? 16 : 0;   // B frag: lane<16 K 0..15, lane>=16 K 16..31
  const int nl   = lane & 15;
  const int mb   = m0 + (hi ? 8 : 0);

  const _Float16* As[3]   = {A1, A2, A3};
  const _Float16* Bs[3]   = {B1, B2, B3};
  const int       Ks[3]   = {K1, K2, K3};
  const int       ldbs[3] = {ldb1, ldb2, ldb3};

  auto store_tile = [&](const v8f& acc, int i) {
    const int nn = n0 + 16 * i + nl;
    const float bv = bias ? bias[nn] : 0.0f;
    #pragma unroll
    for (int v = 0; v < 8; ++v) {
      const float val = acc[v] + bv;
      C[(size_t)(mb + v) * N + nn] = val;
      if (C16) C16[(size_t)(mb + v) * N + nn] = (_Float16)val;
    }
  };

  if (n0 + 64 <= N) {
    // ---------------- fast path: all 4 tiles, straight-line ----------------
    v8f acc0 = {}, acc1 = {}, acc2 = {}, acc3 = {};
    #pragma unroll
    for (int p = 0; p < 3; ++p) {
      const int K = Ks[p];
      if (K <= 0) continue;
      const int ldb = ldbs[p];
      const _Float16* Arow = As[p] + (size_t)(m0 + r) * K + aoff;
      const _Float16* Brow = Bs[p] + (size_t)(n0 + r) * ldb + boff;
      const size_t bstep = (size_t)16 * ldb;
      #pragma unroll 2
      for (int k0 = 0; k0 < K; k0 += 32) {
        v8h alo = *(const v8h*)(Arow + k0);
        v8h ahi = *(const v8h*)(Arow + k0 + 8);
        v16h a  = __builtin_shufflevector(alo, ahi,
                    0,1,2,3,4,5,6,7,8,9,10,11,12,13,14,15);
        __builtin_prefetch(Arow + k0 + 32, 0, 3);       // global_prefetch_b8
        v16h b0 = *(const v16h*)(Brow + k0);
        v16h b1 = *(const v16h*)(Brow + bstep + k0);
        v16h b2 = *(const v16h*)(Brow + 2 * bstep + k0);
        v16h b3 = *(const v16h*)(Brow + 3 * bstep + k0);
        acc0 = WMMA_F16(a, b0, acc0);
        acc1 = WMMA_F16(a, b1, acc1);
        acc2 = WMMA_F16(a, b2, acc2);
        acc3 = WMMA_F16(a, b3, acc3);
      }
    }
    store_tile(acc0, 0);
    store_tile(acc1, 1);
    store_tile(acc2, 2);
    store_tile(acc3, 3);
  } else {
    // ---------------- boundary path (rare: last vocab block) ----------------
    for (int i = 0; i < 4; ++i) {
      const int ns = n0 + 16 * i;
      if (ns + 16 > N) break;
      v8f acc = {};
      for (int p = 0; p < 3; ++p) {
        const int K = Ks[p];
        if (K <= 0) continue;
        const int ldb = ldbs[p];
        const _Float16* Arow = As[p] + (size_t)(m0 + r) * K + aoff;
        const _Float16* Brow = Bs[p] + (size_t)(ns + r) * ldb + boff;
        for (int k0 = 0; k0 < K; k0 += 32) {
          v8h alo = *(const v8h*)(Arow + k0);
          v8h ahi = *(const v8h*)(Arow + k0 + 8);
          v16h a  = __builtin_shufflevector(alo, ahi,
                      0,1,2,3,4,5,6,7,8,9,10,11,12,13,14,15);
          v16h b = *(const v16h*)(Brow + k0);
          acc = WMMA_F16(a, b, acc);
        }
      }
      store_tile(acc, i);
    }
  }
}

// ---------------- f32 -> f16 weight convert with column slice + pad ----------------
__global__ void cvt_f16_pad(const float* __restrict__ src, _Float16* __restrict__ dst,
                            int rows, int scols, int soff, int ccopy, int dcols)
{
  const int total = rows * dcols;
  for (int i = blockIdx.x * blockDim.x + threadIdx.x; i < total;
       i += gridDim.x * blockDim.x) {
    const int rr = i / dcols;
    const int c  = i - rr * dcols;
    const float v = (c < ccopy) ? src[(size_t)rr * scols + soff + c] : 0.0f;
    dst[i] = (_Float16)v;
  }
}

// ---------------- embedding gather (f32 table -> padded f16) ----------------
__global__ void gather_f16_pad(const int* __restrict__ tok, const float* __restrict__ emb,
                               _Float16* __restrict__ dst, int T, int dcols)
{
  const int total = T * NB * dcols;
  for (int i = blockIdx.x * blockDim.x + threadIdx.x; i < total;
       i += gridDim.x * blockDim.x) {
    const int c  = i % dcols;
    const int tn = i / dcols;
    const float v = (c < EDIM) ? emb[(size_t)tok[tn] * EDIM + c] : 0.0f;
    dst[i] = (_Float16)v;
  }
}

__device__ __forceinline__ float sigm(float x) { return 1.0f / (1.0f + __expf(-x)); }

// ---------------- encoder LSTM pointwise (both directions) ----------------
__global__ void lstm_update_enc(const float* __restrict__ gF, const float* __restrict__ gB,
                                float* __restrict__ cF, float* __restrict__ cB,
                                _Float16* __restrict__ hF, _Float16* __restrict__ hB,
                                _Float16* __restrict__ encSt, int t)
{
  const int idx = blockIdx.x * blockDim.x + threadIdx.x;
  if (idx >= 2 * NB * HDIM) return;
  const int dir = idx >> 15;
  const int rem = idx & 32767;
  const int n = rem >> 9;
  const int j = rem & 511;
  const float* g = dir ? gB : gF;
  float*       c = dir ? cB : cF;
  const float gi = sigm(g[n * G4H + j]);
  const float gf = sigm(g[n * G4H + HDIM + j]);
  const float gg = tanhf(g[n * G4H + 2 * HDIM + j]);
  const float go = sigm(g[n * G4H + 3 * HDIM + j]);
  const float cn = gf * c[n * HDIM + j] + gi * gg;
  c[n * HDIM + j] = cn;
  const float h = go * tanhf(cn);
  if (dir) {
    hB[n * HDIM + j] = (_Float16)h;
    encSt[(size_t)((S_SRC - 1 - t) * NB + n) * H2 + HDIM + j] = (_Float16)h;
  } else {
    hF[n * HDIM + j] = (_Float16)h;
    encSt[(size_t)(t * NB + n) * H2 + j] = (_Float16)h;
  }
}

// ---------------- decoder LSTM pointwise: writes h into hAll slice ----------------
__global__ void lstm_update_dec(const float* __restrict__ g,
                                float* __restrict__ c, _Float16* __restrict__ hOut)
{
  const int idx = blockIdx.x * blockDim.x + threadIdx.x;
  if (idx >= NB * HDIM) return;
  const int n = idx >> 9;
  const int j = idx & 511;
  const float gi = sigm(g[n * G4H + j]);
  const float gf = sigm(g[n * G4H + HDIM + j]);
  const float gg = tanhf(g[n * G4H + 2 * HDIM + j]);
  const float go = sigm(g[n * G4H + 3 * HDIM + j]);
  const float cn = gf * c[n * HDIM + j] + gi * gg;
  c[n * HDIM + j] = cn;
  hOut[n * HDIM + j] = (_Float16)(go * tanhf(cn));
}

// ---------------- eW[s,n] = enc_states[s,n,:] . We  (precomputed once) ----------------
__global__ void precompute_eW(const _Float16* __restrict__ enc16,
                              const float* __restrict__ energyW, float* __restrict__ eW)
{
  const int i = blockIdx.x * blockDim.x + threadIdx.x;
  if (i >= S_SRC * NB) return;
  const _Float16* row = enc16 + (size_t)i * H2;
  float acc = 0.0f;
  for (int l = 0; l < H2; ++l) acc += (float)row[l] * energyW[HDIM + l];
  eW[i] = acc;
}

// ---------------- attention + context: one block per batch element ----------------
__global__ void attn_ctx(const _Float16* __restrict__ h16, const _Float16* __restrict__ enc16,
                         const float* __restrict__ energyW, const float* __restrict__ energyB,
                         const float* __restrict__ eW, _Float16* __restrict__ ctx)
{
  const int n   = blockIdx.x;
  const int tid = threadIdx.x;     // 128 threads
  __shared__ float red[128];
  __shared__ float w[64];

  float p = 0.0f;
  for (int j = tid; j < HDIM; j += 128)
    p += (float)h16[n * HDIM + j] * energyW[j];
  red[tid] = p;
  __syncthreads();
  for (int s = 64; s > 0; s >>= 1) {
    if (tid < s) red[tid] += red[tid + s];
    __syncthreads();
  }
  const float hWh = red[0];

  if (tid < S_SRC) {
    const float e = hWh + eW[tid * NB + n] + energyB[0];
    w[tid] = e > 0.0f ? e : 0.0f;
  }
  __syncthreads();
  if (tid == 0) {
    float mx = w[0];
    for (int s = 1; s < S_SRC; ++s) mx = fmaxf(mx, w[s]);
    float sum = 0.0f;
    for (int s = 0; s < S_SRC; ++s) { float ex = __expf(w[s] - mx); w[s] = ex; sum += ex; }
    const float inv = 1.0f / sum;
    for (int s = 0; s < S_SRC; ++s) w[s] *= inv;
  }
  __syncthreads();

  for (int l = tid; l < H2; l += 128) {
    float acc = 0.0f;
    for (int s = 0; s < S_SRC; ++s)
      acc += w[s] * (float)enc16[(size_t)(s * NB + n) * H2 + l];
    ctx[(size_t)n * H2 + l] = (_Float16)acc;
  }
}

// =====================================================================
// Host launcher
// =====================================================================
extern "C" void kernel_launch(void* const* d_in, const int* in_sizes, int n_in,
                              void* d_out, int out_size, void* d_ws, size_t ws_size,
                              hipStream_t stream)
{
  (void)in_sizes; (void)n_in; (void)out_size; (void)ws_size;

  const int*   source    = (const int*)  d_in[0];
  const int*   target    = (const int*)  d_in[1];
  const float* enc_emb   = (const float*)d_in[2];
  const float* enc_Wih_f = (const float*)d_in[3];
  const float* enc_Whh_f = (const float*)d_in[4];
  const float* enc_b_f   = (const float*)d_in[5];
  const float* enc_Wih_b = (const float*)d_in[6];
  const float* enc_Whh_b = (const float*)d_in[7];
  const float* enc_b_b   = (const float*)d_in[8];
  const float* fc_hid_W  = (const float*)d_in[9];
  const float* fc_hid_b  = (const float*)d_in[10];
  const float* fc_cell_W = (const float*)d_in[11];
  const float* fc_cell_b = (const float*)d_in[12];
  const float* dec_emb   = (const float*)d_in[13];
  const float* dec_Wih   = (const float*)d_in[14];
  const float* dec_Whh   = (const float*)d_in[15];
  const float* dec_b     = (const float*)d_in[16];
  const float* energy_W  = (const float*)d_in[17];
  const float* energy_b  = (const float*)d_in[18];
  const float* fc_W      = (const float*)d_in[19];
  const float* fc_b      = (const float*)d_in[20];
  float* out = (float*)d_out;

  // ---------- workspace bump allocator ----------
  size_t off = 0;
  auto alloc = [&](size_t bytes) -> void* {
    void* p = (char*)d_ws + off;
    off = (off + bytes + 255) & ~(size_t)255;
    return p;
  };
  _Float16* wWihF  = (_Float16*)alloc((size_t)G4H * EPAD * 2);
  _Float16* wWhhF  = (_Float16*)alloc((size_t)G4H * HDIM * 2);
  _Float16* wWihB  = (_Float16*)alloc((size_t)G4H * EPAD * 2);
  _Float16* wWhhB  = (_Float16*)alloc((size_t)G4H * HDIM * 2);
  _Float16* wFcHid = (_Float16*)alloc((size_t)HDIM * H2 * 2);
  _Float16* wFcCel = (_Float16*)alloc((size_t)HDIM * H2 * 2);
  _Float16* wWihC  = (_Float16*)alloc((size_t)G4H * H2 * 2);
  _Float16* wWihE  = (_Float16*)alloc((size_t)G4H * EPAD * 2);
  _Float16* wWhhD  = (_Float16*)alloc((size_t)G4H * HDIM * 2);
  _Float16* wFcW   = (_Float16*)alloc((size_t)VOCAB * HDIM * 2);
  _Float16* xEnc   = (_Float16*)alloc((size_t)S_SRC * NB * EPAD * 2);
  _Float16* xDec   = (_Float16*)alloc((size_t)(S_TGT - 1) * NB * EPAD * 2);
  _Float16* encSt  = (_Float16*)alloc((size_t)S_SRC * NB * H2 * 2);
  // hAll: slice 0 = bridge hidden, slice t+1 = decoder h after step t
  _Float16* hAll   = (_Float16*)alloc((size_t)S_TGT * NB * HDIM * 2);
  _Float16* h16f   = (_Float16*)alloc((size_t)NB * HDIM * 2);
  _Float16* h16b   = (_Float16*)alloc((size_t)NB * HDIM * 2);
  float*    cF     = (float*)   alloc((size_t)NB * HDIM * 4);
  float*    cB     = (float*)   alloc((size_t)NB * HDIM * 4);
  float*    cD     = (float*)   alloc((size_t)NB * HDIM * 4);
  float*    gatesF = (float*)   alloc((size_t)NB * G4H * 4);
  float*    gatesB = (float*)   alloc((size_t)NB * G4H * 4);
  float*    gatesD = (float*)   alloc((size_t)NB * G4H * 4);
  _Float16* ctx16  = (_Float16*)alloc((size_t)NB * H2 * 2);
  float*    eW     = (float*)   alloc((size_t)S_SRC * NB * 4);
  float*    tmpF   = (float*)   alloc((size_t)NB * HDIM * 4);

  // ---------- weight conversion (f32 -> padded f16) ----------
  #define CVT(src, dst, rows, scols, soff, ccopy, dcols)                              \
    cvt_f16_pad<<<256, 256, 0, stream>>>((src), (dst), (rows), (scols), (soff),       \
                                         (ccopy), (dcols))
  CVT(enc_Wih_f, wWihF,  G4H,  EDIM,      0,    EDIM, EPAD);
  CVT(enc_Whh_f, wWhhF,  G4H,  HDIM,      0,    HDIM, HDIM);
  CVT(enc_Wih_b, wWihB,  G4H,  EDIM,      0,    EDIM, EPAD);
  CVT(enc_Whh_b, wWhhB,  G4H,  HDIM,      0,    HDIM, HDIM);
  CVT(fc_hid_W,  wFcHid, HDIM, H2,        0,    H2,   H2);
  CVT(fc_cell_W, wFcCel, HDIM, H2,        0,    H2,   H2);
  CVT(dec_Wih,   wWihC,  G4H,  H2 + EDIM, 0,    H2,   H2);    // context part
  CVT(dec_Wih,   wWihE,  G4H,  H2 + EDIM, H2,   EDIM, EPAD);  // embedding part
  CVT(dec_Whh,   wWhhD,  G4H,  HDIM,      0,    HDIM, HDIM);
  CVT(fc_W,      wFcW,   VOCAB,HDIM,      0,    HDIM, HDIM);
  #undef CVT

  // ---------- embedding gathers ----------
  gather_f16_pad<<<512, 256, 0, stream>>>(source, enc_emb, xEnc, S_SRC, EPAD);
  gather_f16_pad<<<512, 256, 0, stream>>>(target, dec_emb, xDec, S_TGT - 1, EPAD);

  // ---------- zero init states and t=0 output slab ----------
  hipMemsetAsync(h16f, 0, (size_t)NB * HDIM * 2, stream);
  hipMemsetAsync(h16b, 0, (size_t)NB * HDIM * 2, stream);
  hipMemsetAsync(cF,   0, (size_t)NB * HDIM * 4, stream);
  hipMemsetAsync(cB,   0, (size_t)NB * HDIM * 4, stream);
  hipMemsetAsync(out,  0, (size_t)NB * VOCAB * 4, stream);

  const dim3 gblk(128);

  // ---------- encoder: 50 bidirectional steps ----------
  for (int t = 0; t < S_SRC; ++t) {
    gemm3_wmma<<<dim3(G4H / 64, 1), gblk, 0, stream>>>(
        xEnc + (size_t)t * NB * EPAD, wWihF, EPAD, EPAD,
        h16f, wWhhF, HDIM, HDIM,
        nullptr, nullptr, 0, 0,
        enc_b_f, gatesF, nullptr, G4H);
    gemm3_wmma<<<dim3(G4H / 64, 1), gblk, 0, stream>>>(
        xEnc + (size_t)(S_SRC - 1 - t) * NB * EPAD, wWihB, EPAD, EPAD,
        h16b, wWhhB, HDIM, HDIM,
        nullptr, nullptr, 0, 0,
        enc_b_b, gatesB, nullptr, G4H);
    lstm_update_enc<<<(2 * NB * HDIM + 255) / 256, 256, 0, stream>>>(
        gatesF, gatesB, cF, cB, h16f, h16b, encSt, t);
  }

  // ---------- bridge: hidden (-> hAll slice 0) / cell ----------
  gemm3_wmma<<<dim3(HDIM / 64, 1), gblk, 0, stream>>>(
      h16f, wFcHid, HDIM, H2,
      h16b, wFcHid + HDIM, HDIM, H2,
      nullptr, nullptr, 0, 0,
      fc_hid_b, tmpF, hAll, HDIM);
  gemm3_wmma<<<dim3(HDIM / 64, 1), gblk, 0, stream>>>(
      h16f, wFcCel, HDIM, H2,
      h16b, wFcCel + HDIM, HDIM, H2,
      nullptr, nullptr, 0, 0,
      fc_cell_b, cD, nullptr, HDIM);

  // ---------- attention key term (once) ----------
  precompute_eW<<<(S_SRC * NB + 255) / 256, 256, 0, stream>>>(encSt, energy_W, eW);

  // ---------- decoder recurrence: 49 steps (no vocab GEMM inside!) ----------
  for (int t = 0; t < S_TGT - 1; ++t) {
    const _Float16* hCur = hAll + (size_t)t * NB * HDIM;
    attn_ctx<<<NB, 128, 0, stream>>>(hCur, encSt, energy_W, energy_b, eW, ctx16);
    gemm3_wmma<<<dim3(G4H / 64, 1), gblk, 0, stream>>>(
        ctx16, wWihC, H2, H2,
        xDec + (size_t)t * NB * EPAD, wWihE, EPAD, EPAD,
        hCur, wWhhD, HDIM, HDIM,
        dec_b, gatesD, nullptr, G4H);
    lstm_update_dec<<<(NB * HDIM + 255) / 256, 256, 0, stream>>>(
        gatesD, cD, hAll + (size_t)(t + 1) * NB * HDIM);
  }

  // ---------- deferred vocab projection: ONE big GEMM ----------
  // A = hAll slices 1..49  (M = 49*64 = 3136, K = 512)
  // C = out rows 64..3199  (contiguous: row t*64+n -> preds[t+1][n])
  gemm3_wmma<<<dim3((VOCAB + 63) / 64, (S_TGT - 1) * NB / 64), gblk, 0, stream>>>(
      hAll + (size_t)NB * HDIM, wFcW, HDIM, HDIM,
      nullptr, nullptr, 0, 0,
      nullptr, nullptr, 0, 0,
      fc_b, out + (size_t)NB * VOCAB, nullptr, VOCAB);
}
